// CrossAttention_11570641895747
// MI455X (gfx1250) — compile-verified, matrix operand order checked by reading, not run on hardware
//
#include <hip/hip_runtime.h>
#include <hip/hip_bf16.h>

typedef __attribute__((ext_vector_type(16))) __bf16 v16bf;
typedef __attribute__((ext_vector_type(8)))  __bf16 v8bf;
typedef __attribute__((ext_vector_type(8)))  float  v8f;

#define DIMD 128
#define HEADS 4
#define DH 32
#define SCALE 0.17677669529663687f  // 32^-0.5
#define NVIEW 6
#define QTOK 2304   // 48*48
#define KTOK 576    // 24*24
#define BATCH 2
#define FPAD 4      // float LDS row pad (banks)
#define HPAD 8      // bf16 LDS row pad (keeps 16B row alignment)

// DPP controls for a 16-lane butterfly (xor1, xor2, xor4==j^7, xor8==j^15)
#define DPP_QUAD_XOR1   0xB1   // quad_perm [1,0,3,2]
#define DPP_QUAD_XOR2   0x4E   // quad_perm [2,3,0,1]
#define DPP_HALF_MIRROR 0x141  // j -> j^7 within 8
#define DPP_ROW_MIRROR  0x140  // j -> j^15 within 16

template <int CTRL>
__device__ __forceinline__ float dpp_xchg(float x) {
  int yi = __builtin_amdgcn_update_dpp(0, __float_as_int(x), CTRL, 0xF, 0xF, true);
  return __int_as_float(yi);
}
// max over the 16 lanes of each half-wave (pure VALU, no LDS)
__device__ __forceinline__ float red16_max(float x) {
  x = fmaxf(x, dpp_xchg<DPP_QUAD_XOR1>(x));
  x = fmaxf(x, dpp_xchg<DPP_QUAD_XOR2>(x));
  x = fmaxf(x, dpp_xchg<DPP_HALF_MIRROR>(x));
  x = fmaxf(x, dpp_xchg<DPP_ROW_MIRROR>(x));
  return x;
}
__device__ __forceinline__ float red16_sum(float x) {
  x += dpp_xchg<DPP_QUAD_XOR1>(x);
  x += dpp_xchg<DPP_QUAD_XOR2>(x);
  x += dpp_xchg<DPP_HALF_MIRROR>(x);
  x += dpp_xchg<DPP_ROW_MIRROR>(x);
  return x;
}

__device__ __forceinline__ v8f wmma_bf(v16bf a, v16bf b, v8f c) {
  return __builtin_amdgcn_wmma_f32_16x16x32_bf16(false, a, false, b, (short)0, c,
                                                 false, false);
}

// Per-lane fragment elements are two contiguous 8xbf16 (16B) runs -> b128 loads.
// lanes 0-15 hold K = {0..7, 16..23}, lanes 16-31 hold K = {8..15, 24..31}
__device__ __forceinline__ v16bf load_frag_rowmajor(const __bf16* rowstart, int lane) {
  int b0 = (lane & 16) ? 8 : 0;
  v8bf lo = *(const v8bf*)(rowstart + b0);
  v8bf hi = *(const v8bf*)(rowstart + b0 + 16);
  v16bf r;
#pragma unroll
  for (int i = 0; i < 8; i++) { r[i] = lo[i]; r[i + 8] = hi[i]; }
  return r;
}

// Packed-weight B fragment: lane's 16 elements are fully contiguous (32B).
__device__ __forceinline__ v16bf load_frag_packed(const __bf16* Wpk, int kc, int CT,
                                                  int ct, int lane) {
  const __bf16* p = Wpk + (((size_t)(kc * CT + ct) * 32 + lane) << 4);
  v8bf lo = *(const v8bf*)p;
  v8bf hi = *(const v8bf*)(p + 8);
  v16bf r;
#pragma unroll
  for (int i = 0; i < 8; i++) { r[i] = lo[i]; r[i + 8] = hi[i]; }
  return r;
}

// fp32 (K x ncols) row-major -> bf16 packed in B-fragment order:
// dst[((kc*CT + ct)*32 + lane)*16 + i] = W[(kc*32 + kmap(lane,i)) * ncols + ct*16 + (lane&15)]
__global__ void pack_w_kernel(const float* __restrict__ src, __bf16* __restrict__ dst,
                              int K, int ncols) {
  int idx = blockIdx.x * blockDim.x + threadIdx.x;
  if (idx >= K * ncols) return;
  int i = idx & 15;
  int lane = (idx >> 4) & 31;
  int rest = idx >> 9;
  int CT = ncols >> 4;
  int ct = rest % CT;
  int kc = rest / CT;
  int col = ct * 16 + (lane & 15);
  int k = kc * 32 + ((lane & 16) ? 8 : 0) + ((i < 8) ? i : (i + 8));
  dst[idx] = (__bf16)src[(size_t)k * ncols + col];
}

// ---------------- LN + projection (x is (BN, 128, S) channel-strided) -------------
// TRANSPOSE == 0: y[token*128 + col]         (token-major, for Q/K)
// TRANSPOSE == 1: y[(bn*128 + col)*S + key]  (dh-major per view, for V)
template <int S, int TRANSPOSE>
__global__ __launch_bounds__(256)
void ln_proj_kernel(const float* __restrict__ x,
                    const float* __restrict__ gamma, const float* __restrict__ beta,
                    const __bf16* __restrict__ Wpk, const float* __restrict__ bias,
                    __bf16* __restrict__ y) {
  __shared__ float xs[16][DIMD + FPAD];
  __shared__ float rsum[16][16];
  __shared__ float rsum2[16][16];
  __shared__ float mean_s[16], rstd_s[16];
  __shared__ __bf16 abf[16][DIMD + HPAD];

  const int tid = threadIdx.x;
  const int t0 = blockIdx.x * 16;
  const int bn0 = t0 / S;              // S % 16 == 0: whole tile shares bn
  const int sp0 = t0 - bn0 * S;

  // token-fast staging: 16 consecutive lanes read 16 consecutive spatial floats
  for (int idx = tid; idx < 16 * DIMD; idx += 256) {
    int tok = idx & 15, c = idx >> 4;
    xs[tok][c] = x[(size_t)(bn0 * DIMD + c) * S + sp0 + tok];
  }
  __syncthreads();
  {
    int tok = tid >> 4, l16 = tid & 15;
    float s = 0.f, s2 = 0.f;
#pragma unroll
    for (int j = 0; j < 8; j++) {
      float v = xs[tok][l16 * 8 + j];
      s += v; s2 += v * v;
    }
    rsum[tok][l16] = s; rsum2[tok][l16] = s2;
  }
  __syncthreads();
  if (tid < 16) {
    float s = 0.f, s2 = 0.f;
    for (int j = 0; j < 16; j++) { s += rsum[tid][j]; s2 += rsum2[tid][j]; }
    float m = s / 128.f;
    mean_s[tid] = m;
    rstd_s[tid] = rsqrtf(s2 / 128.f - m * m + 1e-5f);
  }
  __syncthreads();
  for (int idx = tid; idx < 16 * DIMD; idx += 256) {
    int tok = idx >> 7, c = idx & 127;
    abf[tok][c] = (__bf16)((xs[tok][c] - mean_s[tok]) * rstd_s[tok] * gamma[c] + beta[c]);
  }
  __syncthreads();

  const int lane = tid & 31;
  const int wv = tid >> 5;                 // 8 waves -> 8 x 16 output cols
  const int row = lane & 15;
  const int coln = wv * 16 + (lane & 15);
  const int mbase = (lane & 16) ? 8 : 0;
  v8f acc = {};
#pragma unroll
  for (int kc = 0; kc < 4; kc++) {
    v16bf a = load_frag_rowmajor(&abf[row][kc * 32], lane);
    v16bf b = load_frag_packed(Wpk, kc, 8, wv, lane);
    acc = wmma_bf(a, b, acc);
  }
  float bb = bias[coln];
  if (TRANSPOSE == 0) {
#pragma unroll
    for (int r = 0; r < 8; r++)
      y[(size_t)(t0 + mbase + r) * DIMD + coln] = (__bf16)(acc[r] + bb);
  } else {
    v8bf pk;
#pragma unroll
    for (int r = 0; r < 8; r++) pk[r] = (__bf16)(acc[r] + bb);
    *(v8bf*)(y + ((size_t)(bn0 * DIMD + coln)) * S + sp0 + mbase) = pk;
  }
}

// ---------------- Flash attention: one wave per (bh, 16-query tile) --------------
__global__ __launch_bounds__(32)
void attn_kernel(const __bf16* __restrict__ qp, const __bf16* __restrict__ kp,
                 const __bf16* __restrict__ vt, __bf16* __restrict__ z1) {
  const int lane = threadIdx.x;
  const int q0 = blockIdx.x * 16;
  const int bh = blockIdx.y;
  const int b = bh >> 2, h = bh & 3;

  __shared__ __bf16 p_lds[16][32 + HPAD];

  v8f o0 = {}, o1 = {};
  float mrow[8], lrow[8];
#pragma unroll
  for (int r = 0; r < 8; r++) { mrow[r] = -1e30f; lrow[r] = 0.f; }

  const int row = lane & 15;
  const int col = lane & 15;
  const int mbase = (lane & 16) ? 8 : 0;

  for (int n = 0; n < NVIEW; n++) {
    // query A-frag for this view: 16 q-rows x DH=32 (K of the WMMA)
    const __bf16* qrow = qp + ((size_t)((b * NVIEW + n) * QTOK + q0 + row)) * DIMD + h * DH;
    v16bf aq = load_frag_rowmajor(qrow, lane);

    const __bf16* kbp = kp + (size_t)(b * NVIEW + n) * KTOK * DIMD + h * DH;
    // vt: (bn, 128 dh-major, 576 keys)
    const __bf16* vtp = vt + ((size_t)(b * NVIEW + n) * DIMD + h * DH) * KTOK;

    for (int kb = 0; kb < KTOK; kb += 32) {
      // prefetch next key/value tiles into near caches (sequential stream)
      __builtin_prefetch(kbp + (size_t)(kb + 32 + col) * DIMD, 0, 3);
      __builtin_prefetch(vtp + (size_t)col * KTOK + kb + 32, 0, 3);

      // K^T B-frags: lane = key column, elements run along DH (row-major in kp)
      v16bf bk0 = load_frag_rowmajor(kbp + (size_t)(kb + col) * DIMD, lane);
      v16bf bk1 = load_frag_rowmajor(kbp + (size_t)(kb + 16 + col) * DIMD, lane);
      v8f zc = {};
      v8f s0 = wmma_bf(aq, bk0, zc);
      v8f s1 = wmma_bf(aq, bk1, zc);

#pragma unroll
      for (int r = 0; r < 8; r++) {
        float a0 = s0[r] * SCALE;
        float a1 = s1[r] * SCALE;
        float mx = red16_max(fmaxf(a0, a1));          // DPP butterfly, no LDS
        float mnew = fmaxf(mrow[r], mx);
        float alpha = __expf(mrow[r] - mnew);
        float p0 = __expf(a0 - mnew);
        float p1 = __expf(a1 - mnew);
        float sum = red16_sum(p0 + p1);               // DPP butterfly, no LDS
        lrow[r] = lrow[r] * alpha + sum;
        mrow[r] = mnew;
        o0[r] *= alpha;
        o1[r] *= alpha;
        p_lds[mbase + r][col]      = (__bf16)p0;      // C-layout -> LDS (row-major)
        p_lds[mbase + r][col + 16] = (__bf16)p1;
      }
      // P A-frag (intra-wave LDS ordering: loads after stores are in order)
      v16bf ap = load_frag_rowmajor(&p_lds[row][0], lane);
      // V B-frags from dh-major vt: lane = dh column, elements contiguous along keys
      v16bf bv0 = load_frag_rowmajor(vtp + (size_t)col * KTOK + kb, lane);
      v16bf bv1 = load_frag_rowmajor(vtp + (size_t)(col + 16) * KTOK + kb, lane);
      o0 = wmma_bf(ap, bv0, o0);
      o1 = wmma_bf(ap, bv1, o1);
    }
  }
#pragma unroll
  for (int r = 0; r < 8; r++) {
    float inv = 1.f / lrow[r];
    size_t base = ((size_t)(b * QTOK + q0 + mbase + r)) * DIMD + h * DH;
    z1[base + col]      = (__bf16)(o0[r] * inv);
    z1[base + col + 16] = (__bf16)(o1[r] * inv);
  }
}

// shared LN helper (zs already populated + synced)
__device__ __forceinline__ void ln_stats(float (*zs)[DIMD], float* mean_s, float* rstd_s,
                                         float (*rsum)[16], float (*rsum2)[16], int tid) {
  int tok = tid >> 4, l16 = tid & 15;
  float s = 0.f, s2 = 0.f;
#pragma unroll
  for (int j = 0; j < 8; j++) {
    float v = zs[tok][l16 * 8 + j];
    s += v; s2 += v * v;
  }
  rsum[tok][l16] = s; rsum2[tok][l16] = s2;
  __syncthreads();
  if (tid < 16) {
    float a = 0.f, a2 = 0.f;
    for (int j = 0; j < 16; j++) { a += rsum[tid][j]; a2 += rsum2[tid][j]; }
    float m = a / 128.f;
    mean_s[tid] = m;
    rstd_s[tid] = rsqrtf(a2 / 128.f - m * m + 1e-5f);
  }
  __syncthreads();
}

// ------------- z2 = LN_pre(z1 @ Wp + bp + skip) ----------------------------------
__global__ __launch_bounds__(256)
void proj_skip_ln_kernel(const __bf16* __restrict__ z1, const __bf16* __restrict__ Wppk,
                         const float* __restrict__ bp, const float* __restrict__ skip,
                         const float* __restrict__ pre_g, const float* __restrict__ pre_b,
                         float* __restrict__ z2) {
  __shared__ float zs[16][DIMD];
  __shared__ float rsum[16][16], rsum2[16][16];
  __shared__ float mean_s[16], rstd_s[16];
  const int tid = threadIdx.x;
  const int t0 = blockIdx.x * 16;
  const int bidx = t0 / QTOK, sp0 = t0 - bidx * QTOK;   // QTOK % 16 == 0
  const int lane = tid & 31, wv = tid >> 5;
  const int row = lane & 15;
  const int coln = wv * 16 + (lane & 15);
  const int mbase = (lane & 16) ? 8 : 0;

  v8f acc = {};
#pragma unroll
  for (int kc = 0; kc < 4; kc++) {
    v16bf a = load_frag_rowmajor(z1 + (size_t)(t0 + row) * DIMD + kc * 32, lane);
    v16bf b = load_frag_packed(Wppk, kc, 8, wv, lane);
    acc = wmma_bf(a, b, acc);
  }
  float bb = bp[coln];
  const float* skp = skip + (size_t)(bidx * DIMD + coln) * QTOK + sp0 + mbase;
#pragma unroll
  for (int r = 0; r < 8; r++)
    zs[mbase + r][coln] = acc[r] + bb + skp[r];
  __syncthreads();
  ln_stats(zs, mean_s, rstd_s, rsum, rsum2, tid);
  for (int idx = tid; idx < 16 * DIMD; idx += 256) {   // c-fast: z2 token-major
    int tok = idx >> 7, c = idx & 127;
    z2[(size_t)(t0 + tok) * DIMD + c] =
        (zs[tok][c] - mean_s[tok]) * rstd_s[tok] * pre_g[c] + pre_b[c];
  }
}

// ------------- h = gelu(z2 @ W1 + b1), 128 -> 256 --------------------------------
__global__ __launch_bounds__(256)
void mlp1_kernel(const float* __restrict__ z2, const __bf16* __restrict__ W1pk,
                 const float* __restrict__ b1, __bf16* __restrict__ hb) {
  __shared__ __bf16 abf[16][DIMD + HPAD];
  const int tid = threadIdx.x;
  const int t0 = blockIdx.x * 16;
  for (int idx = tid; idx < 16 * DIMD; idx += 256) {
    int tok = idx >> 7, c = idx & 127;
    abf[tok][c] = (__bf16)z2[(size_t)(t0 + tok) * DIMD + c];
  }
  __syncthreads();
  const int lane = tid & 31, wv = tid >> 5;
  const int row = lane & 15;
  const int mbase = (lane & 16) ? 8 : 0;
#pragma unroll
  for (int ct = 0; ct < 2; ct++) {           // 8 waves x 2 = 16 col tiles of 256
    const int ctile = wv * 2 + ct;
    const int coln = ctile * 16 + (lane & 15);
    v8f acc = {};
#pragma unroll
    for (int kc = 0; kc < 4; kc++) {
      v16bf a = load_frag_rowmajor(&abf[row][kc * 32], lane);
      v16bf b = load_frag_packed(W1pk, kc, 16, ctile, lane);
      acc = wmma_bf(a, b, acc);
    }
    float bb = b1[coln];
#pragma unroll
    for (int r = 0; r < 8; r++) {
      float x = acc[r] + bb;
      float g = 0.5f * x * (1.f + erff(x * 0.70710678118654752f));
      hb[(size_t)(t0 + mbase + r) * 256 + coln] = (__bf16)g;
    }
  }
}

// ------------- out = LN_post(z2 + h @ W2 + b2), scattered to (B,128,2304) --------
__global__ __launch_bounds__(256)
void mlp2_ln_out_kernel(const __bf16* __restrict__ hb, const __bf16* __restrict__ W2pk,
                        const float* __restrict__ b2, const float* __restrict__ z2,
                        const float* __restrict__ post_g, const float* __restrict__ post_b,
                        float* __restrict__ out) {
  __shared__ float zs[16][DIMD];
  __shared__ float rsum[16][16], rsum2[16][16];
  __shared__ float mean_s[16], rstd_s[16];
  const int tid = threadIdx.x;
  const int t0 = blockIdx.x * 16;
  const int bidx = t0 / QTOK, sp0 = t0 - bidx * QTOK;
  const int lane = tid & 31, wv = tid >> 5;
  const int row = lane & 15;
  const int coln = wv * 16 + (lane & 15);
  const int mbase = (lane & 16) ? 8 : 0;

  v8f acc = {};
#pragma unroll
  for (int kc = 0; kc < 8; kc++) {           // K = 256
    v16bf a = load_frag_rowmajor(hb + (size_t)(t0 + row) * 256 + kc * 32, lane);
    v16bf b = load_frag_packed(W2pk, kc, 8, wv, lane);
    acc = wmma_bf(a, b, acc);
  }
  float bb = b2[coln];
#pragma unroll
  for (int r = 0; r < 8; r++) {
    int m = mbase + r;
    zs[m][coln] = acc[r] + bb + z2[(size_t)(t0 + m) * DIMD + coln];
  }
  __syncthreads();
  ln_stats(zs, mean_s, rstd_s, rsum, rsum2, tid);
  // token-fast: 16 consecutive lanes write 16 consecutive spatial floats
  for (int idx = tid; idx < 16 * DIMD; idx += 256) {
    int tok = idx & 15, c = idx >> 4;
    out[(size_t)(bidx * DIMD + c) * QTOK + sp0 + tok] =
        (zs[tok][c] - mean_s[tok]) * rstd_s[tok] * post_g[c] + post_b[c];
  }
}

extern "C" void kernel_launch(void* const* d_in, const int* in_sizes, int n_in,
                              void* d_out, int out_size, void* d_ws, size_t ws_size,
                              hipStream_t stream) {
  const float* q      = (const float*)d_in[0];
  const float* k      = (const float*)d_in[1];
  const float* v      = (const float*)d_in[2];
  const float* skip   = (const float*)d_in[3];
  const float* lnq_g  = (const float*)d_in[4];
  const float* lnq_b  = (const float*)d_in[5];
  const float* Wq     = (const float*)d_in[6];
  const float* bq     = (const float*)d_in[7];
  const float* lnk_g  = (const float*)d_in[8];
  const float* lnk_b  = (const float*)d_in[9];
  const float* Wk     = (const float*)d_in[10];
  const float* bk     = (const float*)d_in[11];
  const float* lnv_g  = (const float*)d_in[12];
  const float* lnv_b  = (const float*)d_in[13];
  const float* Wv     = (const float*)d_in[14];
  const float* bv     = (const float*)d_in[15];
  const float* Wp     = (const float*)d_in[16];
  const float* bp     = (const float*)d_in[17];
  const float* pre_g  = (const float*)d_in[18];
  const float* pre_b  = (const float*)d_in[19];
  const float* W1     = (const float*)d_in[20];
  const float* b1     = (const float*)d_in[21];
  const float* W2     = (const float*)d_in[22];
  const float* b2     = (const float*)d_in[23];
  const float* post_g = (const float*)d_in[24];
  const float* post_b = (const float*)d_in[25];
  float* out = (float*)d_out;

  char* ws = (char*)d_ws;
  size_t off = 0;
  auto alloc = [&](size_t bytes) {
    void* p = ws + off;
    off += (bytes + 255) & ~(size_t)255;
    return p;
  };
  const int NQ = BATCH * NVIEW * QTOK;   // 27648
  const int NK = BATCH * NVIEW * KTOK;   // 6912
  const int NT = BATCH * QTOK;           // 4608

  __bf16* Wq_p = (__bf16*)alloc(16384 * 2);
  __bf16* Wk_p = (__bf16*)alloc(16384 * 2);
  __bf16* Wv_p = (__bf16*)alloc(16384 * 2);
  __bf16* Wp_p = (__bf16*)alloc(16384 * 2);
  __bf16* W1_p = (__bf16*)alloc(32768 * 2);
  __bf16* W2_p = (__bf16*)alloc(32768 * 2);
  __bf16* qp_b = (__bf16*)alloc((size_t)NQ * DIMD * 2);
  __bf16* kp_b = (__bf16*)alloc((size_t)NK * DIMD * 2);
  __bf16* vt_b = (__bf16*)alloc((size_t)NK * DIMD * 2);  // dh-major per view
  __bf16* z1_b = (__bf16*)alloc((size_t)NT * DIMD * 2);
  float*  z2   = (float*) alloc((size_t)NT * DIMD * 4);
  __bf16* h_b  = (__bf16*)alloc((size_t)NT * 256 * 2);

  pack_w_kernel<<<64,  256, 0, stream>>>(Wq, Wq_p, 128, 128);
  pack_w_kernel<<<64,  256, 0, stream>>>(Wk, Wk_p, 128, 128);
  pack_w_kernel<<<64,  256, 0, stream>>>(Wv, Wv_p, 128, 128);
  pack_w_kernel<<<64,  256, 0, stream>>>(Wp, Wp_p, 128, 128);
  pack_w_kernel<<<128, 256, 0, stream>>>(W1, W1_p, 128, 256);
  pack_w_kernel<<<128, 256, 0, stream>>>(W2, W2_p, 256, 128);

  ln_proj_kernel<QTOK, 0><<<NQ / 16, 256, 0, stream>>>(q, lnq_g, lnq_b, Wq_p, bq, qp_b);
  ln_proj_kernel<KTOK, 0><<<NK / 16, 256, 0, stream>>>(k, lnk_g, lnk_b, Wk_p, bk, kp_b);
  ln_proj_kernel<KTOK, 1><<<NK / 16, 256, 0, stream>>>(v, lnv_g, lnv_b, Wv_p, bv, vt_b);

  dim3 agrid(QTOK / 16, BATCH * HEADS);
  attn_kernel<<<agrid, 32, 0, stream>>>(qp_b, kp_b, vt_b, z1_b);

  proj_skip_ln_kernel<<<NT / 16, 256, 0, stream>>>(z1_b, Wp_p, bp, skip, pre_g, pre_b, z2);
  mlp1_kernel<<<NT / 16, 256, 0, stream>>>(z2, W1_p, b1, h_b);
  mlp2_ln_out_kernel<<<NT / 16, 256, 0, stream>>>(h_b, W2_p, b2, z2, post_g, post_b, out);
}
